// Decoder_34041910788720
// MI455X (gfx1250) — matrix operand order repaired and checked by reading, not compiled, not run on hardware
//
#include <hip/hip_runtime.h>
#include <hip/hip_bf16.h>
#include <stdint.h>

#define BATCH  128
#define HID    1024
#define VOC    32000
#define TLEN   16
#define BOS_ID 1

typedef __attribute__((ext_vector_type(16))) __bf16 v16bf;
typedef __attribute__((ext_vector_type(8)))  __bf16 v8bf;
typedef __attribute__((ext_vector_type(8)))  float  v8f;

__device__ __forceinline__ uint16_t f2bf_bits(float f) {
    uint32_t u = __builtin_bit_cast(uint32_t, f);
    u += 0x7FFFu + ((u >> 16) & 1u);            // round-to-nearest-even
    return (uint16_t)(u >> 16);
}

// -----------------------------------------------------------------------------
// C(128 x N) = A(128 x K, bf16) @ W^T + bias.   W is (N x K) row-major, read as
// bf16 (pre-converted, L2-resident on MI455X) or fp32 (fallback, cvt in LDS).
// grid.x = N/32, block = 256 = 8 wave32s; wave w owns rows [16w,16w+16) and the
// full 32-col tile (2 accumulators -> 2 WMMAs per A fragment per k-step).
// B tile (32K x 32N) double-buffered in LDS, software-pipelined: the stage of
// tile i+1 is issued before the fragment loads / WMMAs of tile i, and the
// last tile is peeled so the steady-state loop is branch-free.
// -----------------------------------------------------------------------------
template <bool USE_BF>
__global__ __launch_bounds__(256) void gemm_bias_wmma(
    const uint16_t* __restrict__ Abf,   // (128 x K) bf16
    const void*     __restrict__ Wp,    // (N x K) bf16 (USE_BF) or fp32
    const float*    __restrict__ bias,
    float*          __restrict__ Cf,    // fp32 out (or null)
    uint16_t*       __restrict__ Cb,    // bf16 out (or null)
    int K, int N)
{
    __shared__ __bf16 Bt[2][32 * 40];   // 32 N-cols x 32 K, stride 40 (16B rows)

    const int t     = threadIdx.x;
    const int wave  = t >> 5;
    const int lane  = t & 31;
    const int g     = lane >> 4;        // half-wave group (ISA fragment layout)
    const int nloc  = lane & 15;
    const int ncol0 = blockIdx.x * 32;

    // staging coords: thread t moves 4 consecutive bf16 of the 32x32 tile
    const int sn = t >> 3;              // 0..31 : N within tile
    const int sk = (t & 7) * 4;         // 0..28 : K within tile
    const size_t wrow = (size_t)(ncol0 + sn) * (size_t)K;
    const uint16_t* Wb = (const uint16_t*)Wp;
    const float*    Wf = (const float*)Wp;

    auto stage = [&](int k0, int buf) {
        if (USE_BF) {
            uint2 p = *(const uint2*)(Wb + wrow + k0 + sk);
            *(uint2*)&Bt[buf][sn * 40 + sk] = p;
            __builtin_prefetch(Wb + wrow + k0 + 32 + sk, 0, 1);  // spec. prefetch: OOB dropped
        } else {
            float4 w4 = *(const float4*)(Wf + wrow + k0 + sk);
            uint2 p;
            p.x = (uint32_t)f2bf_bits(w4.x) | ((uint32_t)f2bf_bits(w4.y) << 16);
            p.y = (uint32_t)f2bf_bits(w4.z) | ((uint32_t)f2bf_bits(w4.w) << 16);
            *(uint2*)&Bt[buf][sn * 40 + sk] = p;
            __builtin_prefetch(Wf + wrow + k0 + 32 + sk, 0, 1);
        }
    };

    v8f acc0 = {}, acc1 = {};
    const uint16_t* arow = Abf + (size_t)(wave * 16 + nloc) * (size_t)K;

    auto compute = [&](int k0, int buf) {
        // A fragment (16x32 bf16): two aligned b128 loads, no conversion VALU.
        // lanes g=0: K 0..15 ; g=1: K 16..31
        v8bf alo = *(const v8bf*)(arow + k0 + 16 * g);
        v8bf ahi = *(const v8bf*)(arow + k0 + 16 * g + 8);
        v16bf a  = __builtin_shufflevector(alo, ahi,
                     0,1,2,3,4,5,6,7,8,9,10,11,12,13,14,15);

        // B fragments (32x16 bf16 each) from LDS: lane holds col (lane&15),
        // K range 16*g..16*g+15 -> two b128 ds loads per fragment.
        const __bf16* bp0 = &Bt[buf][nloc * 40 + 16 * g];
        const __bf16* bp1 = &Bt[buf][(16 + nloc) * 40 + 16 * g];
        v16bf b0 = __builtin_shufflevector(*(const v8bf*)bp0, *(const v8bf*)(bp0 + 8),
                     0,1,2,3,4,5,6,7,8,9,10,11,12,13,14,15);
        v16bf b1 = __builtin_shufflevector(*(const v8bf*)bp1, *(const v8bf*)(bp1 + 8),
                     0,1,2,3,4,5,6,7,8,9,10,11,12,13,14,15);

        acc0 = __builtin_amdgcn_wmma_f32_16x16x32_bf16(
                   false, a, false, b0, (short)0, acc0, false, false);
        acc1 = __builtin_amdgcn_wmma_f32_16x16x32_bf16(
                   false, a, false, b1, (short)0, acc1, false, false);
    };

    const int nk = K >> 5;              // K is always a multiple of 32 here
    stage(0, 0);
    __syncthreads();
    int buf = 0;
    for (int i = 0; i < nk - 1; ++i) {  // branch-free steady state
        stage((i + 1) << 5, buf ^ 1);
        compute(i << 5, buf);
        __syncthreads();
        buf ^= 1;
    }
    compute((nk - 1) << 5, buf);        // peeled last tile

    const int col0 = ncol0 + nloc;
    const int col1 = col0 + 16;
    const float bv0 = bias[col0], bv1 = bias[col1];
    #pragma unroll
    for (int i = 0; i < 8; ++i) {
        int m = wave * 16 + 8 * g + i;          // C/D VGPR i -> row 8g+i (ISA layout)
        size_t o0 = (size_t)m * (size_t)N + col0;
        size_t o1 = (size_t)m * (size_t)N + col1;
        if (Cf) {
            Cf[o0] = acc0[i] + bv0;
            Cf[o1] = acc1[i] + bv1;
        } else {
            Cb[o0] = f2bf_bits(acc0[i] + bv0);
            Cb[o1] = f2bf_bits(acc1[i] + bv1);
        }
    }
}

// fused GRU gates: h = (1-z)*n + z*h ; also writes bf16 mirror for next GEMMs
__global__ void gru_combine(const float* __restrict__ gi,
                            const float* __restrict__ gh,
                            float* __restrict__ h,
                            uint16_t* __restrict__ hbf, int n)
{
    int idx = blockIdx.x * blockDim.x + threadIdx.x;
    if (idx >= n) return;
    int b = idx / HID, j = idx - b * HID;
    const float* gib = gi + (size_t)b * 3 * HID;
    const float* ghb = gh + (size_t)b * 3 * HID;
    float r  = 1.f / (1.f + __expf(-(gib[j]           + ghb[j])));
    float z  = 1.f / (1.f + __expf(-(gib[HID + j]     + ghb[HID + j])));
    float nn = tanhf(gib[2 * HID + j] + r * ghb[2 * HID + j]);
    float hv = (1.f - z) * nn + z * h[idx];
    h[idx]   = hv;
    hbf[idx] = f2bf_bits(hv);
}

// softmax over the BATCH axis (reference quirk). One vocab column per thread,
// consecutive threads -> consecutive columns (coalesced). Emits bf16 probs
// (they are only ever consumed as a WMMA A operand next step).
__global__ void softmax_batch(const float* __restrict__ logits,
                              uint16_t* __restrict__ probs_bf)
{
    int col = blockIdx.x * blockDim.x + threadIdx.x;
    if (col >= VOC) return;
    float m = -1e30f;
    for (int b = 0; b < BATCH; ++b)
        m = fmaxf(m, logits[(size_t)b * VOC + col]);
    float s = 0.f;
    for (int b = 0; b < BATCH; ++b)
        s += __expf(logits[(size_t)b * VOC + col] - m);
    float inv = 1.f / s;
    for (int b = 0; b < BATCH; ++b)
        probs_bf[(size_t)b * VOC + col] =
            f2bf_bits(__expf(logits[(size_t)b * VOC + col] - m) * inv);
}

__global__ void init_probs_bf(uint16_t* __restrict__ p, int n)
{
    int idx = blockIdx.x * blockDim.x + threadIdx.x;
    if (idx >= n) return;
    p[idx] = ((idx % VOC) == BOS_ID) ? (uint16_t)0x3F80 : (uint16_t)0;  // bf16 1.0
}

// seed h0/h1 (fp32 working copy + bf16 mirror) from `hidden`
__global__ void init_h(const float* __restrict__ hidden,
                       float* __restrict__ h, uint16_t* __restrict__ hbf, int n)
{
    int idx = blockIdx.x * blockDim.x + threadIdx.x;
    if (idx >= n) return;
    float v = hidden[idx];
    h[idx]   = v;
    hbf[idx] = f2bf_bits(v);
}

__global__ void cvt_f32_bf16(const float* __restrict__ s,
                             uint16_t* __restrict__ d, int n4)
{
    int i = blockIdx.x * blockDim.x + threadIdx.x;
    if (i >= n4) return;
    float4 v = ((const float4*)s)[i];
    uint2 r;
    r.x = (uint32_t)f2bf_bits(v.x) | ((uint32_t)f2bf_bits(v.y) << 16);
    r.y = (uint32_t)f2bf_bits(v.z) | ((uint32_t)f2bf_bits(v.w) << 16);
    ((uint2*)d)[i] = r;
}

extern "C" void kernel_launch(void* const* d_in, const int* in_sizes, int n_in,
                              void* d_out, int out_size, void* d_ws, size_t ws_size,
                              hipStream_t stream)
{
    (void)in_sizes; (void)n_in; (void)out_size;
    const float* hidden = (const float*)d_in[0];
    const float* w_in   = (const float*)d_in[1];
    const float* b_in   = (const float*)d_in[2];
    const float* W_ih0  = (const float*)d_in[3];
    const float* W_hh0  = (const float*)d_in[4];
    const float* b_ih0  = (const float*)d_in[5];
    const float* b_hh0  = (const float*)d_in[6];
    const float* W_ih1  = (const float*)d_in[7];
    const float* W_hh1  = (const float*)d_in[8];
    const float* b_ih1  = (const float*)d_in[9];
    const float* b_hh1  = (const float*)d_in[10];
    const float* w_out  = (const float*)d_in[11];
    const float* b_out  = (const float*)d_in[12];
    float* out = (float*)d_out;

    char* ws = (char*)d_ws;
    size_t off = 0;
    auto take = [&](size_t bytes) {
        size_t o = off; off = (off + bytes + 255) & ~(size_t)255; return o;
    };
    // bf16 activations (WMMA A operands)
    uint16_t* probsb = (uint16_t*)(ws + take((size_t)BATCH * VOC * 2));
    uint16_t* xb     = (uint16_t*)(ws + take((size_t)BATCH * HID * 2));
    uint16_t* h01b   = (uint16_t*)(ws + take((size_t)2 * BATCH * HID * 2));
    // fp32 intermediates
    float* gi  = (float*)(ws + take((size_t)BATCH * 3 * HID * 4));
    float* gh  = (float*)(ws + take((size_t)BATCH * 3 * HID * 4));
    float* h01 = (float*)(ws + take((size_t)2 * BATCH * HID * 4));
    float*    h0  = h01;
    float*    h1  = h01  + (size_t)BATCH * HID;
    uint16_t* h0b = h01b;
    uint16_t* h1b = h01b + (size_t)BATCH * HID;
    // bf16 weight cache (156 MB total -> fits MI455X 192 MB L2 for all 16 steps)
    uint16_t* winb  = (uint16_t*)(ws + take((size_t)HID * VOC * 2));
    uint16_t* wih0b = (uint16_t*)(ws + take((size_t)3 * HID * HID * 2));
    uint16_t* whh0b = (uint16_t*)(ws + take((size_t)3 * HID * HID * 2));
    uint16_t* wih1b = (uint16_t*)(ws + take((size_t)3 * HID * HID * 2));
    uint16_t* whh1b = (uint16_t*)(ws + take((size_t)3 * HID * HID * 2));
    uint16_t* woutb = (uint16_t*)(ws + take((size_t)VOC * HID * 2));
    const bool use_bf = (ws_size >= off);   // deterministic: ws_size is fixed

    { int n = 2 * BATCH * HID;
      init_h<<<(n + 255) / 256, 256, 0, stream>>>(hidden, h01, h01b, n); }
    { int n = BATCH * VOC;
      init_probs_bf<<<(n + 255) / 256, 256, 0, stream>>>(probsb, n); }

    if (use_bf) {
        auto cv = [&](const float* s, uint16_t* d, size_t n) {
            int n4 = (int)(n / 4);
            cvt_f32_bf16<<<(n4 + 255) / 256, 256, 0, stream>>>(s, d, n4);
        };
        cv(w_in,  winb,  (size_t)HID * VOC);
        cv(W_ih0, wih0b, (size_t)3 * HID * HID);
        cv(W_hh0, whh0b, (size_t)3 * HID * HID);
        cv(W_ih1, wih1b, (size_t)3 * HID * HID);
        cv(W_hh1, whh1b, (size_t)3 * HID * HID);
        cv(w_out, woutb, (size_t)VOC * HID);
    }

    auto gemm = [&](const uint16_t* A, const float* Wf, const uint16_t* Wb,
                    const float* bias, float* Cf, uint16_t* Cb, int K, int N) {
        if (use_bf)
            gemm_bias_wmma<true><<<N / 32, 256, 0, stream>>>(
                A, (const void*)Wb, bias, Cf, Cb, K, N);
        else
            gemm_bias_wmma<false><<<N / 32, 256, 0, stream>>>(
                A, (const void*)Wf, bias, Cf, Cb, K, N);
    };

    const int nBH = BATCH * HID;
    for (int t = 0; t < TLEN; ++t) {
        float* logits = out + (size_t)t * BATCH * VOC;
        gemm(probsb, w_in,  winb,  b_in,  nullptr, xb,  VOC, HID);     // emb_in (bf16 out)
        gemm(xb,     W_ih0, wih0b, b_ih0, gi, nullptr, HID, 3 * HID);  // GRU0 input gates
        gemm(h0b,    W_hh0, whh0b, b_hh0, gh, nullptr, HID, 3 * HID);  // GRU0 hidden gates
        gru_combine<<<(nBH + 255) / 256, 256, 0, stream>>>(gi, gh, h0, h0b, nBH);
        gemm(h0b,    W_ih1, wih1b, b_ih1, gi, nullptr, HID, 3 * HID);  // GRU1 input gates
        gemm(h1b,    W_hh1, whh1b, b_hh1, gh, nullptr, HID, 3 * HID);  // GRU1 hidden gates
        gru_combine<<<(nBH + 255) / 256, 256, 0, stream>>>(gi, gh, h1, h1b, nBH);
        gemm(h1b,    w_out, woutb, b_out, logits, nullptr, HID, VOC);  // emb_out
        softmax_batch<<<VOC / 256, 256, 0, stream>>>(logits, probsb);
    }
}
